// ActivationUnit_69054484185514
// MI455X (gfx1250) — compile-verified, matrix operand order checked by reading, not run on hardware
//
#include <hip/hip_runtime.h>
#include <math.h>

// ---------------------------------------------------------------------------
// DIN-style "activation unit" attention for MI455X (gfx1250, wave32, WMMA).
//   din = [q, k, q-k, q*k] @ W1  ==  q@(W1a+W1c) + k@(W1b-W1c) + (q*k)@W1d
// -> per-(b,l) GEMM has K=128 only; q-term folded into the bias (per b).
// Layer1/2 run on v_wmma_f32_16x16x32_bf16; keys staged once in LDS (bf16)
// and reused for the softmax-weighted sum (single HBM pass over 210 MB).
// Sigmoid uses v_rcp_f32 (not IEEE div) -> 3 VALU ops per element.
// ---------------------------------------------------------------------------

typedef __bf16 bf16;
typedef __attribute__((ext_vector_type(16))) __bf16 v16bf;
typedef __attribute__((ext_vector_type(8)))  float  v8f;
typedef __attribute__((ext_vector_type(4)))  unsigned int u32x4;

#define D_   64
#define L_   200
#define H1_  80
#define H2_  40
#define NT_  13          // ceil(200/16) row tiles
#define LP_  208         // padded L

// LDS pitches in BYTES (rows 16B-aligned; pitches chosen so 16 lanes reading
// row m hit distinct banks: 272B=68 dw -> 4m mod 64; 208B=52 dw -> 52m mod 64)
#define A_PITCH    272   // 136 bf16 (A operand rows, K=128 used)
#define W1T_PITCH  272   // W1' transposed [80][128] bf16
#define H_PITCH    208   // h1 rows [16][96] bf16 (K padded 80->96)
#define W2T_PITCH  208   // W2 transposed+padded [48][96] bf16
#define H2_PITCH   41    // h2 rows [16][40] f32, odd pitch -> conflict free
#define KEY_PITCH  128   // keys rows [64] bf16

// LDS layout (bytes)
#define W1T_OFF   0
#define W2T_OFF   (W1T_OFF + 80 * W1T_PITCH)        // 21760
#define KEYS_OFF  (W2T_OFF + 48 * W2T_PITCH)        // 31744
#define WSCR_OFF  (KEYS_OFF + LP_ * KEY_PITCH)      // 58368
#define WAVE_SCR  6144   // per wave: Abuf 4352 | then Hbuf 3328 + h2buf 2624
#define QF_OFF    (WSCR_OFF + 4 * WAVE_SCR)         // 82944
#define QW1_OFF   (QF_OFF + 256)                    // 83200 (80 f32)
#define B2_OFF    (QW1_OFF + 320)                   // 83520 (40 f32, pad)
#define W3_OFF    (B2_OFF + 192)                    // 83712 (40 f32, pad)
#define SC_OFF    (W3_OFF + 192)                    // 83904 (208 f32)
#define RED_OFF   (SC_OFF + 832)                    // 84736 (128 f32)
#define SMEM_BYTES (RED_OFF + 512)                  // 85248 (<320KB WGP LDS)

union Frag { v16bf v; u32x4 q[2]; };

__device__ __forceinline__ unsigned pack2(float a, float b) {
    union { __bf16 h[2]; unsigned u; } t;
    t.h[0] = (__bf16)a; t.h[1] = (__bf16)b;
    return t.u;
}
// Fast sigmoid: v_exp_f32 + v_add + v_rcp_f32 (no IEEE div sequence).
__device__ __forceinline__ float sigmoidf_(float x) {
    return __builtin_amdgcn_rcpf(1.0f + __expf(-x));
}

__global__ __launch_bounds__(128)
void actunit_kernel(const float* __restrict__ qg,
                    const float* __restrict__ keysg,
                    const unsigned char* __restrict__ maskg, // jnp.bool_ = 1B
                    const float* __restrict__ W1, const float* __restrict__ b1,
                    const float* __restrict__ W2, const float* __restrict__ b2,
                    const float* __restrict__ W3, const float* __restrict__ b3,
                    float* __restrict__ outg)
{
    extern __shared__ char smem[];
    const int tid  = threadIdx.x;
    const int lane = tid & 31;
    const int w    = tid >> 5;
    const int b    = blockIdx.x;

    float* qf   = (float*)(smem + QF_OFF);
    float* qw1  = (float*)(smem + QW1_OFF);
    float* b2s  = (float*)(smem + B2_OFF);
    float* w3s  = (float*)(smem + W3_OFF);
    float* scor = (float*)(smem + SC_OFF);
    float* red  = (float*)(smem + RED_OFF);
    bf16*  w1t  = (bf16*)(smem + W1T_OFF);
    bf16*  w2t  = (bf16*)(smem + W2T_OFF);

    // ---------------- setup: stage q, biases, bf16 weight tiles ------------
    if (tid < D_) qf[tid] = qg[(size_t)b * D_ + tid];
    if (tid < H2_) { b2s[tid] = b2[tid]; w3s[tid] = W3[tid]; }
    __syncthreads();

    // qW1[n] = b1[n] + sum_d q[d]*(W1a+W1c)[d][n]   (constant over L)
    for (int n = tid; n < H1_; n += 128) {
        float s = b1[n];
        for (int d = 0; d < D_; ++d)
            s += qf[d] * (W1[d * H1_ + n] + W1[(128 + d) * H1_ + n]);
        qw1[n] = s;
    }
    // W1' transposed: [n][k], k<64 -> (W1b-W1c), k>=64 -> W1d
    for (int idx = tid; idx < 128 * H1_; idx += 128) {
        int k = idx / H1_, n = idx - k * H1_;
        float v = (k < 64) ? (W1[(64 + k) * H1_ + n] - W1[(128 + k) * H1_ + n])
                           : W1[(192 + (k - 64)) * H1_ + n];
        w1t[n * (W1T_PITCH / 2) + k] = (bf16)v;
    }
    // W2 transposed+padded: [48][104] bf16, zeros outside 40x80
    for (int idx = tid; idx < 48 * (W2T_PITCH / 2); idx += 128) {
        int n = idx / (W2T_PITCH / 2), k = idx - n * (W2T_PITCH / 2);
        float v = (n < H2_ && k < H1_) ? W2[k * H2_ + n] : 0.0f;
        w2t[n * (W2T_PITCH / 2) + k] = (bf16)v;
    }
    const float b3v = b3[0];
    __syncthreads();

    // ---------------- pass 1: per-wave 16-row tiles ------------------------
    char* wscr = smem + WSCR_OFF + w * WAVE_SCR;
    const int m  = lane & 15;        // WMMA M-row (A) / N-col (B,C)
    const int kh = lane >> 4;        // half-lane K offset (+8) per ISA layout

    for (int t = w; t < NT_; t += 4) {
        const int r0 = t * 16;

        // Load 16x64 f32 keys, build A = [k | q*k] bf16, cache keys (bf16)
        #pragma unroll
        for (int j = 0; j < 8; ++j) {
            int idx = j * 32 + lane;
            int row = idx >> 4, c4 = (idx & 15) * 4;
            int l = r0 + row;
            float4 kv = make_float4(0.f, 0.f, 0.f, 0.f);
            if (l < L_)
                kv = *(const float4*)(keysg + ((size_t)b * L_ + l) * D_ + c4);
            if (l + 64 < L_)   // prefetch next tile owned by this wave
                __builtin_prefetch(keysg + ((size_t)b * L_ + l + 64) * D_ + c4, 0, 0);
            float q0 = qf[c4], q1 = qf[c4 + 1], q2 = qf[c4 + 2], q3 = qf[c4 + 3];
            unsigned pk0 = pack2(kv.x, kv.y),          pk1 = pack2(kv.z, kv.w);
            unsigned pq0 = pack2(kv.x * q0, kv.y * q1), pq1 = pack2(kv.z * q2, kv.w * q3);
            unsigned* arow = (unsigned*)(wscr + row * A_PITCH);
            arow[c4 / 2]            = pk0;  arow[c4 / 2 + 1]        = pk1;
            arow[(64 + c4) / 2]     = pq0;  arow[(64 + c4) / 2 + 1] = pq1;
            unsigned* krow = (unsigned*)(smem + KEYS_OFF + (size_t)(r0 + row) * KEY_PITCH);
            krow[c4 / 2] = pk0;  krow[c4 / 2 + 1] = pk1;
        }
        asm volatile("s_wait_dscnt 0x0" ::: "memory");

        // ---- layer 1: 16x128 @ 128x80, 4 K-steps x 5 N-tiles = 20 WMMA ----
        Frag a1[4];
        #pragma unroll
        for (int ks = 0; ks < 4; ++ks) {
            const char* p = wscr + m * A_PITCH + (ks * 32 + kh * 8) * 2;
            a1[ks].q[0] = *(const u32x4*)p;
            a1[ks].q[1] = *(const u32x4*)(p + 32);
        }
        #pragma unroll
        for (int nt = 0; nt < 5; ++nt) {
            int n = nt * 16 + m;
            float c0 = qw1[n];
            v8f acc = { c0, c0, c0, c0, c0, c0, c0, c0 };
            #pragma unroll
            for (int ks = 0; ks < 4; ++ks) {
                Frag bf_;
                const char* p = smem + W1T_OFF + n * W1T_PITCH + (ks * 32 + kh * 8) * 2;
                bf_.q[0] = *(const u32x4*)p;
                bf_.q[1] = *(const u32x4*)(p + 32);
                acc = __builtin_amdgcn_wmma_f32_16x16x32_bf16(
                        false, a1[ks].v, false, bf_.v, (short)0, acc, false, false);
            }
            // sigmoid -> h1 bf16 rows (reuse wscr; A already in registers)
            #pragma unroll
            for (int i = 0; i < 8; ++i) {
                int mr = i + 8 * kh;   // C layout: VGPR i -> M=i / M=8+i
                ((bf16*)(wscr + mr * H_PITCH))[n] = (bf16)sigmoidf_(acc[i]);
            }
        }
        // zero K-pad columns 80..95 of h1
        #pragma unroll
        for (int i = 0; i < 8; ++i)
            ((bf16*)(wscr + (i + 8 * kh) * H_PITCH))[80 + m] = (bf16)0.0f;
        asm volatile("s_wait_dscnt 0x0" ::: "memory");

        // ---- layer 2: 16x96 @ 96x48, 3 K-steps x 3 N-tiles = 9 WMMA -------
        Frag a2[3];
        #pragma unroll
        for (int ks = 0; ks < 3; ++ks) {
            const char* p = wscr + m * H_PITCH + (ks * 32 + kh * 8) * 2;
            a2[ks].q[0] = *(const u32x4*)p;
            a2[ks].q[1] = *(const u32x4*)(p + 32);
        }
        float* h2b = (float*)(wscr + 16 * H_PITCH);   // 16x41 f32
        #pragma unroll
        for (int nt = 0; nt < 3; ++nt) {
            int n = nt * 16 + m;
            float c0 = (n < H2_) ? b2s[n] : 0.0f;
            v8f acc = { c0, c0, c0, c0, c0, c0, c0, c0 };
            #pragma unroll
            for (int ks = 0; ks < 3; ++ks) {
                Frag bf_;
                const char* p = smem + W2T_OFF + n * W2T_PITCH + (ks * 32 + kh * 8) * 2;
                bf_.q[0] = *(const u32x4*)p;
                bf_.q[1] = *(const u32x4*)(p + 32);
                acc = __builtin_amdgcn_wmma_f32_16x16x32_bf16(
                        false, a2[ks].v, false, bf_.v, (short)0, acc, false, false);
            }
            if (n < H2_) {
                #pragma unroll
                for (int i = 0; i < 8; ++i)
                    h2b[(i + 8 * kh) * H2_PITCH + n] = sigmoidf_(acc[i]);
            }
        }
        asm volatile("s_wait_dscnt 0x0" ::: "memory");

        // ---- layer 3 + mask + 1/sqrt(D): lanes 0..15, one row each --------
        if (lane < 16) {
            int l = r0 + lane;
            if (l < L_) {
                float s = b3v;
                const float* hr = h2b + lane * H2_PITCH;
                #pragma unroll 8
                for (int g = 0; g < H2_; ++g) s += hr[g] * w3s[g];
                s *= 0.125f;                              // 1/sqrt(64)
                if (maskg[(size_t)b * L_ + l]) s = -INFINITY;
                scor[l] = s;
            }
        }
    }
    __syncthreads();

    // ---------------- pass 2: softmax over L + attn @ keys (LDS bf16) ------
    float mx = -INFINITY;
    for (int l = tid; l < L_; l += 128) mx = fmaxf(mx, scor[l]);
    red[tid] = mx; __syncthreads();
    for (int s = 64; s > 0; s >>= 1) {
        if (tid < s) red[tid] = fmaxf(red[tid], red[tid + s]);
        __syncthreads();
    }
    const float gmax = red[0]; __syncthreads();

    float psum = 0.0f;
    for (int l = tid; l < L_; l += 128) {
        float p = __expf(scor[l] - gmax);
        scor[l] = p;
        psum += p;
    }
    red[tid] = psum; __syncthreads();
    for (int s = 64; s > 0; s >>= 1) {
        if (tid < s) red[tid] += red[tid + s];
        __syncthreads();
    }
    const float rsum = __builtin_amdgcn_rcpf(red[0]); __syncthreads();

    const int d = tid & 63, g = tid >> 6;     // 2 L-halves x 64 dims
    const bf16* kb = (const bf16*)(smem + KEYS_OFF);
    float accu = 0.0f;
    #pragma unroll 4
    for (int l = g * 100; l < g * 100 + 100; ++l)
        accu += scor[l] * (float)kb[(size_t)l * (KEY_PITCH / 2) + d];
    red[tid] = accu; __syncthreads();
    if (tid < 64)
        outg[(size_t)b * D_ + tid] = (red[tid] + red[tid + 64]) * rsum;
}

// ---------------------------------------------------------------------------
extern "C" void kernel_launch(void* const* d_in, const int* in_sizes, int n_in,
                              void* d_out, int out_size, void* d_ws, size_t ws_size,
                              hipStream_t stream) {
    (void)n_in; (void)out_size; (void)d_ws; (void)ws_size;
    const float*         q    = (const float*)d_in[0];
    const float*         keys = (const float*)d_in[1];
    const unsigned char* msk  = (const unsigned char*)d_in[2]; // jnp.bool_ (1B)
    const float* W1 = (const float*)d_in[3];
    const float* b1 = (const float*)d_in[4];
    const float* W2 = (const float*)d_in[5];
    const float* b2 = (const float*)d_in[6];
    const float* W3 = (const float*)d_in[7];
    const float* b3 = (const float*)d_in[8];
    float* out = (float*)d_out;

    const int nb = in_sizes[0] / D_;   // batch rows (4096)

    // >64KB dynamic LDS: raise the cap (safe under graph capture, no stream op)
    hipFuncSetAttribute(reinterpret_cast<const void*>(actunit_kernel),
                        hipFuncAttributeMaxDynamicSharedMemorySize, SMEM_BYTES);

    actunit_kernel<<<nb, 128, SMEM_BYTES, stream>>>(
        q, keys, msk, W1, b1, W2, b2, W3, b3, out);
}
